// WKVSpatialCayleyRotation_85263690760268
// MI455X (gfx1250) — matrix operand order repaired and checked by reading, not compiled, not run on hardware
//
#include <hip/hip_runtime.h>
#include <hip/hip_bf16.h>

typedef __attribute__((ext_vector_type(16))) _Float16 v16h;
typedef __attribute__((ext_vector_type(8)))  _Float16 v8h;
typedef __attribute__((ext_vector_type(4)))  _Float16 v4h;
typedef __attribute__((ext_vector_type(8)))  float    v8f;
typedef __attribute__((ext_vector_type(4)))  float    v4f;

// ---------------------------------------------------------------------------
// Kernel 1: batched Cayley transform, one 256-thread block per head matrix.
//   A = 0.5*(P - P^T);  solve (I - A) Y = (I + A)  (Gauss-Jordan, no pivoting
//   needed: symmetric part of I-A is I, SPD).  Then R = Y^T, stored f16
//   row-major into workspace for the WMMA apply kernel.
// ---------------------------------------------------------------------------
__global__ __launch_bounds__(256)
void cayley_kernel(const float* __restrict__ p_row,
                   const float* __restrict__ p_col,
                   _Float16* __restrict__ R1h,
                   _Float16* __restrict__ R2h,
                   int nheads)
{
    __shared__ float Lm[64][65];   // I - A  (padded to dodge bank conflicts)
    __shared__ float Rm[64][65];   // I + A  ->  Y = R^T

    const int h = blockIdx.x;
    const float*   P;
    _Float16*      Rout;
    if (h < nheads) { P = p_row + (size_t)h * 4096;            Rout = R1h + (size_t)h * 4096; }
    else            { P = p_col + (size_t)(h - nheads) * 4096; Rout = R2h + (size_t)(h - nheads) * 4096; }

    const int t = threadIdx.x;

    for (int idx = t; idx < 4096; idx += 256) {
        int r = idx >> 6, c = idx & 63;
        float a = 0.5f * (P[r * 64 + c] - P[c * 64 + r]);
        float e = (r == c) ? 1.0f : 0.0f;
        Lm[r][c] = e - a;
        Rm[r][c] = e + a;
    }
    __syncthreads();

    const int r  = t >> 2;   // row this thread owns (0..63)
    const int cg = t & 3;    // column group (0..3) over 128 augmented columns

    for (int k = 0; k < 64; ++k) {
        if (r == k) {                     // scale pivot row (4 lanes, lockstep)
            float inv = 1.0f / Lm[k][k];
            for (int j = cg * 32; j < cg * 32 + 32; ++j) {
                if (j < 64) Lm[k][j] *= inv;
                else        Rm[k][j - 64] *= inv;
            }
        }
        __syncthreads();
        if (r != k) {                     // eliminate column k everywhere else
            float f = Lm[r][k];
            for (int j = cg * 32; j < cg * 32 + 32; ++j) {
                if (j < 64) Lm[r][j]      -= f * Lm[k][j];
                else        Rm[r][j - 64] -= f * Rm[k][j - 64];
            }
        }
        __syncthreads();
    }

    // Rm holds Y = R^T; emit R row-major in f16:  R[i][j] = Y[j][i]
    for (int idx = t; idx < 4096; idx += 256) {
        int i = idx >> 6, j = idx & 63;
        Rout[idx] = (_Float16)Rm[j][i];
    }
}

// ---------------------------------------------------------------------------
// WMMA operand loaders for v_wmma_f32_16x16x32_f16 (wave32).
//  A (16x32, 16-bit) lane layout:  lanes 0-15 -> row m=lane, K={kb+0..7, kb+16..23}
//                                  lanes 16-31 -> row m=lane-16, K={kb+8..15, kb+24..31}
//  B (32x16, 16-bit) lane layout:  lanes 0-15 -> col n=lane,     K={kb+0..7, kb+16..23}
//                                  lanes 16-31 -> col n=lane-16, K={kb+8..15, kb+24..31}
// All chunk addresses are 16-byte aligned for pitch 64 (global) / 72 (LDS).
// ---------------------------------------------------------------------------
__device__ __forceinline__ v16h load_a16(const _Float16* base, int pitch,
                                         int row, int kb, int hi)
{
    const _Float16* rp = base + row * pitch + kb + hi * 8;
    v8h a0 = *(const v8h*)(rp);
    v8h a1 = *(const v8h*)(rp + 16);
    v16h a;
#pragma unroll
    for (int e = 0; e < 8; ++e) { a[e] = a0[e]; a[8 + e] = a1[e]; }
    return a;
}

__device__ __forceinline__ v16h load_b16(const _Float16* base, int pitch,
                                         int col, int kb, int hi)
{
    const _Float16* cp = base + col * pitch + kb + hi * 8;
    v8h b0 = *(const v8h*)(cp);
    v8h b1 = *(const v8h*)(cp + 16);
    v16h b;
#pragma unroll
    for (int e = 0; e < 8; ++e) { b[e] = b0[e]; b[8 + e] = b1[e]; }
    return b;
}

// ---------------------------------------------------------------------------
// Kernel 2: Out[b,c] = R1[c] * M[b,c] * R2[c]^T  via two chained 64x64 WMMA
// passes.  One 256-thread block (8 waves) per (b,c); each wave owns two 16x16
// tiles per pass.  T = M*R2^T is staged transposed in LDS (f16) so the second
// pass's B operands are contiguous.
// ---------------------------------------------------------------------------
#define LPITCH 72   // LDS pitch in halves: 144B rows, keeps 16B alignment

__global__ __launch_bounds__(256)
void apply_rot_kernel(const float* __restrict__ wkv,
                      const _Float16* __restrict__ R1h,
                      const _Float16* __restrict__ R2h,
                      float* __restrict__ out,
                      int nC)
{
    __shared__ __align__(16) _Float16 Mlds[64 * LPITCH];  // M, f16 row-major
    __shared__ __align__(16) _Float16 Ttl [64 * LPITCH];  // T^T, f16

    const int bc = blockIdx.x;
    const int c  = bc % nC;
    const float*     Mg = wkv + (size_t)bc * 4096;
    float*           Og = out + (size_t)bc * 4096;
    const _Float16*  R1 = R1h + (size_t)c * 4096;
    const _Float16*  R2 = R2h + (size_t)c * 4096;

    const int t = threadIdx.x;

    // ---- stage 0: load M (f32) -> Mlds (f16), coalesced 16B loads ----------
#pragma unroll
    for (int i = 0; i < 4; ++i) {
        int idx = i * 1024 + t * 4;
        v4f m = *(const v4f*)(Mg + idx);
        int r = idx >> 6, cc = idx & 63;
        v4h hh;
        hh[0] = (_Float16)m[0]; hh[1] = (_Float16)m[1];
        hh[2] = (_Float16)m[2]; hh[3] = (_Float16)m[3];
        *(v4h*)(&Mlds[r * LPITCH + cc]) = hh;
    }
    __syncthreads();

    const int lane = t & 31;
    const int wave = t >> 5;
    const int ln   = lane & 15;
    const int hi   = lane >> 4;

    // ---- stage 1: T = M * R2^T  (A = rows of M, B = rows of R2) ------------
#pragma unroll
    for (int q = 0; q < 2; ++q) {
        int tile = wave * 2 + q;
        int ti = tile >> 2, tj = tile & 3;
        v8f acc = {};
#pragma unroll
        for (int kb = 0; kb < 64; kb += 32) {
            v16h a = load_a16(Mlds, LPITCH, ti * 16 + ln, kb, hi);
            v16h b = load_b16(R2,   64,     tj * 16 + ln, kb, hi);
            acc = __builtin_amdgcn_wmma_f32_16x16x32_f16(
                      false, a, false, b, (short)0, acc, false, false);
        }
        // store transposed into LDS: Ttl[col][row], 8 contiguous halves/lane
        v8h hh;
#pragma unroll
        for (int v = 0; v < 8; ++v) hh[v] = (_Float16)acc[v];
        *(v8h*)(&Ttl[(tj * 16 + ln) * LPITCH + ti * 16 + hi * 8]) = hh;
    }
    __syncthreads();

    // ---- stage 2: Out = R1 * T  (A = rows of R1, B = rows of T^T) ----------
#pragma unroll
    for (int q = 0; q < 2; ++q) {
        int tile = wave * 2 + q;
        int ti = tile >> 2, tj = tile & 3;
        v8f acc = {};
#pragma unroll
        for (int kb = 0; kb < 64; kb += 32) {
            v16h a = load_a16(R1,  64,     ti * 16 + ln, kb, hi);
            v16h b = load_b16(Ttl, LPITCH, tj * 16 + ln, kb, hi);
            acc = __builtin_amdgcn_wmma_f32_16x16x32_f16(
                      false, a, false, b, (short)0, acc, false, false);
        }
#pragma unroll
        for (int v = 0; v < 8; ++v)
            Og[(ti * 16 + v + hi * 8) * 64 + tj * 16 + ln] = acc[v];
    }
}

// ---------------------------------------------------------------------------
extern "C" void kernel_launch(void* const* d_in, const int* in_sizes, int n_in,
                              void* d_out, int out_size, void* d_ws, size_t ws_size,
                              hipStream_t stream)
{
    const float* wkv   = (const float*)d_in[0];   // [B,C,64,64] f32
    const float* p_row = (const float*)d_in[1];   // [C,64,64]  f32
    const float* p_col = (const float*)d_in[2];   // [C,64,64]  f32
    float*       out   = (float*)d_out;           // [B,C,64,64] f32

    const int nC  = in_sizes[1] / 4096;           // 32 heads
    const int nBC = in_sizes[0] / 4096;           // 8192 (b,c) matrices

    // workspace: f16 rotation matrices, 2 * nC * 64*64 halves (512 KB @ C=32)
    _Float16* R1h = (_Float16*)d_ws;
    _Float16* R2h = R1h + (size_t)nC * 4096;

    cayley_kernel<<<dim3(2 * nC), dim3(256), 0, stream>>>(p_row, p_col, R1h, R2h, nC);
    apply_rot_kernel<<<dim3(nBC), dim3(256), 0, stream>>>(wkv, R1h, R2h, out, nC);
}
